// SelfAttention_26001732010539
// MI455X (gfx1250) — matrix operand order repaired and checked by reading, not compiled, not run on hardware
//
#include <hip/hip_runtime.h>
#include <hip/hip_bf16.h>
#include <math.h>

// ---- problem constants ----
#define E_DIM 1024
#define S_DIM 2048
#define H_DIM 16
#define HD_DIM 128
#define N_DIM 2

typedef _Float16 f16;
typedef __attribute__((ext_vector_type(16), aligned(16))) _Float16 v16h;
typedef __attribute__((ext_vector_type(8),  aligned(16))) _Float16 v8h;
typedef __attribute__((ext_vector_type(8),  aligned(32))) float    v8f;

// ---------------------------------------------------------------------------
// WMMA helpers (CDNA5 16x16x32 f16 -> f32 accum), wave32 layouts per ISA 7.12.2
// A 16x32: lane<16 holds row M=lane, K {c..c+7, c+16..c+23};
//          lane>=16 holds row M=lane-16, K {c+8..c+15, c+24..c+31}
// B 32x16: lane<16 holds col N=lane, K {0..7, 16..23}; lane>=16 col N=lane-16, K {8..15, 24..31}
// C 16x16: VGPR r = row (r + 8*(lane>=16)), col = lane&15
// ---------------------------------------------------------------------------
__device__ __forceinline__ v8f wmma_f16(v16h a, v16h b, v8f c) {
  return __builtin_amdgcn_wmma_f32_16x16x32_f16(
      /*neg_a=*/false, a, /*neg_b=*/false, b,
      /*c_mod=*/(short)0, c, /*reuse_a=*/false, /*reuse_b=*/false);
}

// row = pointer to this lane's 32-element K chunk start (row of A, length >= 32)
__device__ __forceinline__ v16h load_a_frag(const f16* __restrict__ row, int lane) {
  const int off = (lane & 16) ? 8 : 0;
  v8h lo = *(const v8h*)(row + off);
  v8h hi = *(const v8h*)(row + off + 16);
  v16h a;
#pragma unroll
  for (int i = 0; i < 8; ++i) { a[i] = lo[i]; a[8 + i] = hi[i]; }
  return a;
}

// row = pointer to this lane's column's K-chunk start (contiguous 32 halves)
__device__ __forceinline__ v16h load_b_frag(const f16* __restrict__ row, int lane) {
  const int off = (lane & 16) ? 8 : 0;
  v8h lo = *(const v8h*)(row + off);
  v8h hi = *(const v8h*)(row + off + 16);
  v16h b;
#pragma unroll
  for (int i = 0; i < 8; ++i) { b[i] = lo[i]; b[8 + i] = hi[i]; }
  return b;
}

// ---------------------------------------------------------------------------
// K1: transpose + convert  x[n,e,s] (f32)  ->  xt[n,s,e] (f16)
// ---------------------------------------------------------------------------
__global__ void k_transpose_convert(const float* __restrict__ x, f16* __restrict__ xt) {
  __shared__ float tile[32][33];
  const int n  = blockIdx.z;
  const int e0 = blockIdx.x * 32;
  const int s0 = blockIdx.y * 32;
  const float* xp  = x  + (size_t)n * E_DIM * S_DIM;
  f16*         xto = xt + (size_t)n * S_DIM * E_DIM;
#pragma unroll
  for (int i = threadIdx.y; i < 32; i += 8)
    tile[i][threadIdx.x] = xp[(size_t)(e0 + i) * S_DIM + s0 + threadIdx.x];
  __syncthreads();
#pragma unroll
  for (int i = threadIdx.y; i < 32; i += 8)
    xto[(size_t)(s0 + i) * E_DIM + e0 + threadIdx.x] = (f16)tile[threadIdx.x][i];
}

// K2: elementwise f32 -> f16 convert
__global__ void k_convert_f16(const float* __restrict__ src, f16* __restrict__ dst, int count) {
  int i = blockIdx.x * blockDim.x + threadIdx.x;
  if (i < count) dst[i] = (f16)src[i];
}

// ---------------------------------------------------------------------------
// K3: reduce GEMM  r[n,j,s] = sum_e Wr[j,e] * xt[n,s,e] + br[j]   (f16 out)
// 2x2 register-blocked: one wave computes a 32x32 output tile (4 WMMAs per
// K-chunk from 2 A-frags + 2 B-frags -> 16 FLOP/byte of operand traffic)
// ---------------------------------------------------------------------------
__global__ __launch_bounds__(32) void k_reduce_gemm(
    const f16* __restrict__ wr, const f16* __restrict__ xt,
    const float* __restrict__ br, f16* __restrict__ rbuf) {
  const int lane = threadIdx.x & 31;
  const int jt = blockIdx.x, st = blockIdx.y, n = blockIdx.z;   // 32-wide tiles
  const f16* arow0 = wr + (size_t)(jt * 32 + (lane & 15)) * E_DIM;
  const f16* arow1 = arow0 + (size_t)16 * E_DIM;
  const f16* bbase = xt + (size_t)n * S_DIM * E_DIM;
  const f16* brow0 = bbase + (size_t)(st * 32 + (lane & 15)) * E_DIM;
  const f16* brow1 = brow0 + (size_t)16 * E_DIM;
  v8f acc[2][2] = {{{}, {}}, {{}, {}}};
  for (int kc = 0; kc < E_DIM / 32; ++kc) {
    v16h a0 = load_a_frag(arow0 + kc * 32, lane);
    v16h a1 = load_a_frag(arow1 + kc * 32, lane);
    v16h b0 = load_b_frag(brow0 + kc * 32, lane);
    v16h b1 = load_b_frag(brow1 + kc * 32, lane);
    acc[0][0] = wmma_f16(a0, b0, acc[0][0]);
    acc[0][1] = wmma_f16(a0, b1, acc[0][1]);
    acc[1][0] = wmma_f16(a1, b0, acc[1][0]);
    acc[1][1] = wmma_f16(a1, b1, acc[1][1]);
  }
  const int hi8 = (lane & 16) ? 8 : 0;
  const int cc = lane & 15;
  f16* outp = rbuf + (size_t)n * S_DIM * S_DIM;
#pragma unroll
  for (int mt = 0; mt < 2; ++mt) {
#pragma unroll
    for (int nt = 0; nt < 2; ++nt) {
      const int col = st * 32 + nt * 16 + cc;
#pragma unroll
      for (int r = 0; r < 8; ++r) {
        int j = jt * 32 + mt * 16 + r + hi8;
        outp[(size_t)j * S_DIM + col] = (f16)(acc[mt][nt][r] + br[j]);
      }
    }
  }
}

// ---------------------------------------------------------------------------
// K4: fused Q/K/V projections per head, 2x2 blocked (32 s-rows x 32 out-cols).
// A-fragments (from r) reused across q/k/v and both column tiles.
// q[h,n,s,k], k[h,n,s,k] row-major; v stored TRANSPOSED as vt[h,n,d,s]
// ---------------------------------------------------------------------------
__global__ __launch_bounds__(32) void k_qkv_gemm(
    const f16* __restrict__ rbuf,
    const f16* __restrict__ wq, const f16* __restrict__ wk, const f16* __restrict__ wv,
    const float* __restrict__ bq, const float* __restrict__ bk, const float* __restrict__ bv,
    f16* __restrict__ qo, f16* __restrict__ ko, f16* __restrict__ vto) {
  const int lane = threadIdx.x & 31;
  const int st = blockIdx.x;          // s-tile 0..63 (32 rows)
  const int kt = blockIdx.y;          // out-dim tile 0..3 (32 cols)
  const int hn = blockIdx.z;          // h*2+n, 0..31
  const int h = hn >> 1, n = hn & 1;
  const f16* arow0 = rbuf + (size_t)n * S_DIM * S_DIM
                   + (size_t)(st * 32 + (lane & 15)) * S_DIM + h * HD_DIM;
  const f16* arow1 = arow0 + (size_t)16 * S_DIM;
  const size_t wbase = ((size_t)h * HD_DIM + kt * 32 + (lane & 15)) * HD_DIM;
  const f16* qrow0 = wq + wbase; const f16* qrow1 = qrow0 + (size_t)16 * HD_DIM;
  const f16* krow0 = wk + wbase; const f16* krow1 = krow0 + (size_t)16 * HD_DIM;
  const f16* vrow0 = wv + wbase; const f16* vrow1 = vrow0 + (size_t)16 * HD_DIM;
  v8f cq[2][2] = {{{}, {}}, {{}, {}}};
  v8f ck[2][2] = {{{}, {}}, {{}, {}}};
  v8f cv[2][2] = {{{}, {}}, {{}, {}}};
#pragma unroll
  for (int c = 0; c < HD_DIM / 32; ++c) {
    v16h a0 = load_a_frag(arow0 + c * 32, lane);
    v16h a1 = load_a_frag(arow1 + c * 32, lane);
    v16h q0 = load_b_frag(qrow0 + c * 32, lane);
    v16h q1 = load_b_frag(qrow1 + c * 32, lane);
    cq[0][0] = wmma_f16(a0, q0, cq[0][0]); cq[0][1] = wmma_f16(a0, q1, cq[0][1]);
    cq[1][0] = wmma_f16(a1, q0, cq[1][0]); cq[1][1] = wmma_f16(a1, q1, cq[1][1]);
    v16h k0 = load_b_frag(krow0 + c * 32, lane);
    v16h k1 = load_b_frag(krow1 + c * 32, lane);
    ck[0][0] = wmma_f16(a0, k0, ck[0][0]); ck[0][1] = wmma_f16(a0, k1, ck[0][1]);
    ck[1][0] = wmma_f16(a1, k0, ck[1][0]); ck[1][1] = wmma_f16(a1, k1, ck[1][1]);
    v16h v0 = load_b_frag(vrow0 + c * 32, lane);
    v16h v1 = load_b_frag(vrow1 + c * 32, lane);
    cv[0][0] = wmma_f16(a0, v0, cv[0][0]); cv[0][1] = wmma_f16(a0, v1, cv[0][1]);
    cv[1][0] = wmma_f16(a1, v0, cv[1][0]); cv[1][1] = wmma_f16(a1, v1, cv[1][1]);
  }
  const int hi8 = (lane & 16) ? 8 : 0;
  const int cc = lane & 15;
  const size_t base = (size_t)hn * S_DIM * HD_DIM;
#pragma unroll
  for (int mt = 0; mt < 2; ++mt) {
#pragma unroll
    for (int nt = 0; nt < 2; ++nt) {
      const int col = kt * 32 + nt * 16 + cc;
      const float bqv = bq[h * HD_DIM + col];
      const float bkv = bk[h * HD_DIM + col];
      const float bvv = bv[h * HD_DIM + col];
#pragma unroll
      for (int r = 0; r < 8; ++r) {
        int srow = st * 32 + mt * 16 + r + hi8;
        qo[base + (size_t)srow * HD_DIM + col] = (f16)(cq[mt][nt][r] + bqv);
        ko[base + (size_t)srow * HD_DIM + col] = (f16)(ck[mt][nt][r] + bkv);
        vto[base + (size_t)col * S_DIM + srow] = (f16)(cv[mt][nt][r] + bvv); // transposed
      }
    }
  }
}

// ---------------------------------------------------------------------------
// K5: flash attention. One wave handles 16 query rows x full HD=128.
// Loop over 64 key-blocks of 32 columns: scores via 8 WMMAs, online softmax
// (16-lane shfl reductions stay within correct row group), P repacked via LDS
// into A-layout, PV via 8 WMMAs. No S x S materialization.
// o stored flat in (h,n,s,d) order == the reference's "faithful raw reshape".
// ---------------------------------------------------------------------------
__global__ __launch_bounds__(128) void k_flash_attn(
    const f16* __restrict__ q, const f16* __restrict__ k,
    const f16* __restrict__ vt, f16* __restrict__ obuf) {
  const int lane = threadIdx.x & 31;
  const int wave = threadIdx.x >> 5;
  const int hn = blockIdx.y;
  const int qb = blockIdx.x * 4 + wave;          // 0..127 query block
  __shared__ __attribute__((aligned(64))) f16 plds[4][16][32];

  const size_t base = (size_t)hn * S_DIM * HD_DIM;
  const f16* qrow = q + base + (size_t)(qb * 16 + (lane & 15)) * HD_DIM;
  v16h af[4];
#pragma unroll
  for (int c = 0; c < 4; ++c) af[c] = load_a_frag(qrow + c * 32, lane);

  const v8f vzero = {};
  v8f o[8];
#pragma unroll
  for (int t = 0; t < 8; ++t) o[t] = vzero;
  float m[8], l[8];
#pragma unroll
  for (int r = 0; r < 8; ++r) { m[r] = -INFINITY; l[r] = 0.f; }

  const f16* kbase  = k  + base;
  const f16* vtbase = vt + base;

  for (int kb = 0; kb < S_DIM / 32; ++kb) {
    // ---- scores: 16 x 32 tile = two 16x16 C tiles ----
    v8f s0 = vzero, s1 = vzero;
    const f16* k0 = kbase + (size_t)(kb * 32 + (lane & 15)) * HD_DIM;
    const f16* k1 = k0 + 16 * HD_DIM;
    // prefetch next key-block rows into cache (global_prefetch_b8)
    __builtin_prefetch(k0 + 32 * HD_DIM, 0, 0);
    __builtin_prefetch(k1 + 32 * HD_DIM, 0, 0);
#pragma unroll
    for (int c = 0; c < 4; ++c) {
      s0 = wmma_f16(af[c], load_b_frag(k0 + c * 32, lane), s0);
      s1 = wmma_f16(af[c], load_b_frag(k1 + c * 32, lane), s1);
    }
    // ---- online softmax (no 1/sqrt(d): faithful to reference) ----
    float alpha[8];
#pragma unroll
    for (int r = 0; r < 8; ++r) {
      float mx = fmaxf(s0[r], s1[r]);
#pragma unroll
      for (int off = 1; off < 16; off <<= 1) mx = fmaxf(mx, __shfl_xor(mx, off, 32));
      float mn = fmaxf(m[r], mx);
      alpha[r] = __expf(m[r] - mn);
      m[r] = mn;
      float p0 = __expf(s0[r] - mn);
      float p1 = __expf(s1[r] - mn);
      float rs = p0 + p1;
#pragma unroll
      for (int off = 1; off < 16; off <<= 1) rs += __shfl_xor(rs, off, 32);
      l[r] = l[r] * alpha[r] + rs;
      s0[r] = p0; s1[r] = p1;
    }
#pragma unroll
    for (int t = 0; t < 8; ++t)
#pragma unroll
      for (int r = 0; r < 8; ++r) o[t][r] *= alpha[r];

    // ---- repack P (16x32) into A-fragment layout via LDS ----
    const int hi8 = (lane & 16) ? 8 : 0;
    const int cc = lane & 15;
#pragma unroll
    for (int r = 0; r < 8; ++r) {
      plds[wave][r + hi8][cc]      = (f16)s0[r];
      plds[wave][r + hi8][16 + cc] = (f16)s1[r];
    }
    __syncthreads();
    v16h pa = load_a_frag(&plds[wave][lane & 15][0], lane);
    __syncthreads();

    // ---- O += P x V over the 8 d-tiles ----
#pragma unroll
    for (int t = 0; t < 8; ++t) {
      const f16* vrow = vtbase + (size_t)(t * 16 + (lane & 15)) * S_DIM + kb * 32;
      __builtin_prefetch(vrow + 32, 0, 0);
      o[t] = wmma_f16(pa, load_b_frag(vrow, lane), o[t]);
    }
  }

  // ---- epilogue: normalize and store (h,n,s,d)-contiguous f16 ----
  const int hi8 = (lane & 16) ? 8 : 0;
  const int cc = lane & 15;
  f16* op = obuf + base;
#pragma unroll
  for (int t = 0; t < 8; ++t) {
#pragma unroll
    for (int r = 0; r < 8; ++r) {
      int srow = qb * 16 + r + hi8;
      op[(size_t)srow * HD_DIM + t * 16 + cc] = (f16)(o[t][r] / l[r]);
    }
  }
}

// ---------------------------------------------------------------------------
// K6: output GEMM  out[n,s,e] = sum_d O[n*S+s, d] * Wo[e,d] + bo[e]   (f32 out)
// obuf (h,n,s,d)-flat reinterpreted as (N, S, 2048) row-major (the raw reshape)
// 2x2 register-blocked: 32x32 output tile per wave.
// ---------------------------------------------------------------------------
__global__ __launch_bounds__(32) void k_out_gemm(
    const f16* __restrict__ obuf, const f16* __restrict__ wo,
    const float* __restrict__ bo, float* __restrict__ out) {
  const int lane = threadIdx.x & 31;
  const int st = blockIdx.x, et = blockIdx.y, n = blockIdx.z;   // 32-wide tiles
  const f16* arow0 = obuf + (size_t)n * S_DIM * S_DIM
                   + (size_t)(st * 32 + (lane & 15)) * S_DIM;
  const f16* arow1 = arow0 + (size_t)16 * S_DIM;
  const f16* brow0 = wo + (size_t)(et * 32 + (lane & 15)) * S_DIM;
  const f16* brow1 = brow0 + (size_t)16 * S_DIM;
  v8f acc[2][2] = {{{}, {}}, {{}, {}}};
  for (int kc = 0; kc < S_DIM / 32; ++kc) {
    v16h a0 = load_a_frag(arow0 + kc * 32, lane);
    v16h a1 = load_a_frag(arow1 + kc * 32, lane);
    v16h b0 = load_b_frag(brow0 + kc * 32, lane);
    v16h b1 = load_b_frag(brow1 + kc * 32, lane);
    acc[0][0] = wmma_f16(a0, b0, acc[0][0]);
    acc[0][1] = wmma_f16(a0, b1, acc[0][1]);
    acc[1][0] = wmma_f16(a1, b0, acc[1][0]);
    acc[1][1] = wmma_f16(a1, b1, acc[1][1]);
  }
  const int hi8 = (lane & 16) ? 8 : 0;
  const int cc = lane & 15;
#pragma unroll
  for (int mt = 0; mt < 2; ++mt) {
#pragma unroll
    for (int nt = 0; nt < 2; ++nt) {
      const int col = et * 32 + nt * 16 + cc;
      const float bv = bo[col];
#pragma unroll
      for (int r = 0; r < 8; ++r) {
        int row = st * 32 + mt * 16 + r + hi8;
        out[((size_t)n * S_DIM + row) * E_DIM + col] = acc[mt][nt][r] + bv;
      }
    }
  }
}

// ---------------------------------------------------------------------------
// host launcher
// ---------------------------------------------------------------------------
static inline size_t align_up(size_t v, size_t a) { return (v + a - 1) & ~(a - 1); }

extern "C" void kernel_launch(void* const* d_in, const int* in_sizes, int n_in,
                              void* d_out, int out_size, void* d_ws, size_t ws_size,
                              hipStream_t stream) {
  const float* x   = (const float*)d_in[0];   // (N,E,S)
  const float* W_r = (const float*)d_in[1];   // (S,E)
  const float* b_r = (const float*)d_in[2];   // (S)
  const float* Wq  = (const float*)d_in[3];   // (H,HD,HD)
  const float* bq  = (const float*)d_in[4];   // (H,HD)
  const float* Wk  = (const float*)d_in[5];
  const float* bk  = (const float*)d_in[6];
  const float* Wv  = (const float*)d_in[7];
  const float* bv  = (const float*)d_in[8];
  const float* Wo  = (const float*)d_in[9];   // (E,S)
  const float* bo  = (const float*)d_in[10];  // (E)
  float* out = (float*)d_out;                 // (N,S,E) f32

  // workspace carve-up (f16 buffers)
  char* ws = (char*)d_ws;
  size_t cur = 0;
  auto alloc = [&](size_t elems) {
    size_t off = align_up(cur, 256);
    cur = off + elems * sizeof(f16);
    return (f16*)(ws + off);
  };
  f16* xt   = alloc((size_t)N_DIM * S_DIM * E_DIM);            // 8 MB
  f16* wr16 = alloc((size_t)S_DIM * E_DIM);                    // 4 MB
  f16* wq16 = alloc((size_t)H_DIM * HD_DIM * HD_DIM);
  f16* wk16 = alloc((size_t)H_DIM * HD_DIM * HD_DIM);
  f16* wv16 = alloc((size_t)H_DIM * HD_DIM * HD_DIM);
  f16* wo16 = alloc((size_t)E_DIM * S_DIM);                    // 4 MB
  f16* r16  = alloc((size_t)N_DIM * S_DIM * S_DIM);            // 16 MB
  f16* q16  = alloc((size_t)H_DIM * N_DIM * S_DIM * HD_DIM);   // 16 MB
  f16* k16  = alloc((size_t)H_DIM * N_DIM * S_DIM * HD_DIM);   // 16 MB
  f16* vt16 = alloc((size_t)H_DIM * N_DIM * HD_DIM * S_DIM);   // 16 MB
  f16* o16  = alloc((size_t)H_DIM * N_DIM * S_DIM * HD_DIM);   // 16 MB
  (void)ws_size; (void)n_in; (void)in_sizes; (void)out_size;

  // K1: transpose+convert x
  k_transpose_convert<<<dim3(E_DIM / 32, S_DIM / 32, N_DIM), dim3(32, 8), 0, stream>>>(x, xt);

  // K2: weight converts
  {
    int c1 = S_DIM * E_DIM;          // W_r
    int c2 = H_DIM * HD_DIM * HD_DIM;
    int c3 = E_DIM * S_DIM;          // Wo
    k_convert_f16<<<(c1 + 255) / 256, 256, 0, stream>>>(W_r, wr16, c1);
    k_convert_f16<<<(c2 + 255) / 256, 256, 0, stream>>>(Wq, wq16, c2);
    k_convert_f16<<<(c2 + 255) / 256, 256, 0, stream>>>(Wk, wk16, c2);
    k_convert_f16<<<(c2 + 255) / 256, 256, 0, stream>>>(Wv, wv16, c2);
    k_convert_f16<<<(c3 + 255) / 256, 256, 0, stream>>>(Wo, wo16, c3);
  }

  // K3: reduce GEMM (32x32 tiles per wave)
  k_reduce_gemm<<<dim3(S_DIM / 32, S_DIM / 32, N_DIM), 32, 0, stream>>>(wr16, xt, b_r, r16);

  // K4: Q/K/V projections (V transposed), 32x32 tiles per wave
  k_qkv_gemm<<<dim3(S_DIM / 32, HD_DIM / 32, H_DIM * N_DIM), 32, 0, stream>>>(
      r16, wq16, wk16, wv16, bq, bk, bv, q16, k16, vt16);

  // K5: flash attention
  k_flash_attn<<<dim3(S_DIM / 16 / 4, H_DIM * N_DIM), 128, 0, stream>>>(q16, k16, vt16, o16);

  // K6: output GEMM (32x32 tiles per wave)
  k_out_gemm<<<dim3(S_DIM / 32, E_DIM / 32, N_DIM), 32, 0, stream>>>(o16, wo16, bo, out);
}
